// QLoRABigNet_661424963878
// MI455X (gfx1250) — matrix operand order repaired and checked by reading, not compiled, not run on hardware
//
#include <hip/hip_runtime.h>

#define DIM   1024
#define NL    18
#define R     32
#define BATCH 32768

typedef __attribute__((ext_vector_type(16))) _Float16 v16h;
typedef __attribute__((ext_vector_type(8)))  float    v8f;
typedef __attribute__((ext_vector_type(4)))  _Float16 v4h;

// Load a 32-byte WMMA fragment as two 16B chunks.
__device__ __forceinline__ v16h ld_frag2(const _Float16* lo, const _Float16* hi) {
  union { v16h h; uint4 u[2]; } f;
  f.u[0] = *(const uint4*)lo;
  f.u[1] = *(const uint4*)hi;
  return f.h;
}

__device__ __forceinline__ v8f wmma16(v16h a, v16h b, v8f c) {
  // D = A(16x32 f16) * B(32x16 f16) + C(16x16 f32)
  return __builtin_amdgcn_wmma_f32_16x16x32_f16(false, a, false, b, (short)0, c,
                                                false, false);
}

// Async DMA 16B/lane global -> LDS (no VGPR landing); tracked by ASYNCcnt.
// lds_off is the raw LDS byte offset = low 32 bits of the generic address.
__device__ __forceinline__ void async_cp16(unsigned lds_off,
                                           const _Float16* g) {
  asm volatile("global_load_async_to_lds_b128 %0, %1, off"
               :
               : "v"(lds_off), "v"((unsigned long long)(uintptr_t)g)
               : "memory");
}

__device__ __forceinline__ void wait_async0() {
#if __has_builtin(__builtin_amdgcn_s_wait_asynccnt)
  __builtin_amdgcn_s_wait_asynccnt(0);
#else
  asm volatile("s_wait_asynccnt 0x0" ::: "memory");
#endif
}

// ---------------- prologue: int4-code * group scale -> fp16 weights ---------
__global__ __launch_bounds__(256) void dequant_w(const int* __restrict__ qw,
                                                 const float* __restrict__ sc,
                                                 _Float16* __restrict__ W,
                                                 int n4) {
  int i = blockIdx.x * blockDim.x + threadIdx.x;
  if (i >= n4) return;
  int base   = i * 4;              // flat element index: li*1024*1024 + n*1024 + k
  int k      = base & (DIM - 1);
  int rowAll = base >> 10;         // li*1024 + n
  float s = sc[rowAll * (DIM / 16) + (k >> 4)];
  int4 q = ((const int4*)qw)[i];
  v4h h = {(_Float16)(q.x * s), (_Float16)(q.y * s),
           (_Float16)(q.z * s), (_Float16)(q.w * s)};
  ((v4h*)W)[i] = h;
}

// ---------------- prologue: fp32 -> fp16 (layout preserved) -----------------
__global__ __launch_bounds__(256) void cvt_f32_f16(const float* __restrict__ in,
                                                   _Float16* __restrict__ out,
                                                   int n4) {
  int i = blockIdx.x * blockDim.x + threadIdx.x;
  if (i >= n4) return;
  float4 v = ((const float4*)in)[i];
  v4h h = {(_Float16)v.x, (_Float16)v.y, (_Float16)v.z, (_Float16)v.w};
  ((v4h*)out)[i] = h;
}

// ---------------- lora first stage: t[M,32] = act[M,1024] @ la^T ------------
__global__ __launch_bounds__(256) void lora_t_kernel(
    const _Float16* __restrict__ A, const _Float16* __restrict__ LA,
    _Float16* __restrict__ T) {
  const int tid  = threadIdx.x;
  const int lane = tid & 31;
  const int wave = tid >> 5;
  const int l16  = lane & 15;
  const int hi   = lane >> 4;
  const int m0   = blockIdx.x * 128 + wave * 16;

  v8f acc[2] = {};
  for (int k0 = 0; k0 < DIM; k0 += 32) {
    const _Float16* arow = A + (size_t)(m0 + l16) * DIM + k0;
    v16h af  = ld_frag2(arow + hi * 8, arow + 16 + hi * 8);
    const _Float16* b0 = LA + (size_t)l16 * DIM + k0 + hi * 8;
    const _Float16* b1 = LA + (size_t)(16 + l16) * DIM + k0 + hi * 8;
    v16h bf0 = ld_frag2(b0, b0 + 16);
    v16h bf1 = ld_frag2(b1, b1 + 16);
    acc[0] = wmma16(af, bf0, acc[0]);
    acc[1] = wmma16(af, bf1, acc[1]);
  }
  for (int nt = 0; nt < 2; ++nt)
    for (int i = 0; i < 8; ++i) {
      int m = m0 + hi * 8 + i;
      int r = nt * 16 + l16;
      T[(size_t)m * R + r] = (_Float16)acc[nt][i];
    }
}

// ---------------- main fused GEMM: y = act@W^T + bias + t@lb^T --------------
// Double-buffered LDS filled by async LDS-DMA; one barrier per K-step.
__global__ __launch_bounds__(256) void gemm_qlora(
    const _Float16* __restrict__ A, const _Float16* __restrict__ W,
    const _Float16* __restrict__ LB, const _Float16* __restrict__ T,
    const float* __restrict__ bias, _Float16* __restrict__ Out16,
    float* __restrict__ H, int do_relu, int do_resid) {
  __shared__ _Float16 lA[2][128 * 32];
  __shared__ _Float16 lB[2][128 * 32];

  const int tid   = threadIdx.x;
  const int lane  = tid & 31;
  const int wave  = tid >> 5;
  const int l16   = lane & 15;
  const int hi    = lane >> 4;
  const int m0    = blockIdx.x * 128;
  const int n0    = blockIdx.y * 128;
  const int wm    = (wave >> 2) * 64;   // wave 2x4 grid over 128x128 tile
  const int wn    = (wave & 3) * 32;
  const int lrow  = tid >> 1;           // 2 threads per 64B tile row
  const int lpart = tid & 1;

  const _Float16* gA = A + (size_t)(m0 + lrow) * DIM + lpart * 16;
  const _Float16* gB = W + (size_t)(n0 + lrow) * DIM + lpart * 16;

  v8f acc[4][2] = {};

  // Prologue: issue async fills for buffer 0 (k0 = 0).
  {
    unsigned oa = (unsigned)(uintptr_t)(&lA[0][lrow * 32 + lpart * 16]);
    unsigned ob = (unsigned)(uintptr_t)(&lB[0][lrow * 32 + lpart * 16]);
    async_cp16(oa, gA);
    async_cp16(oa + 16, gA + 8);
    async_cp16(ob, gB);
    async_cp16(ob + 16, gB + 8);
  }

  const int KSTEPS = DIM / 32;
  for (int i = 0; i < KSTEPS; ++i) {
    wait_async0();        // this wave's fills for buf[i&1] are in LDS
    __syncthreads();      // all waves' fills done; prior-iter readers done

    if (i + 1 < KSTEPS) { // issue fills for next K-slice into other buffer
      int nb = (i + 1) & 1;
      int k1 = (i + 1) * 32;
      unsigned oa = (unsigned)(uintptr_t)(&lA[nb][lrow * 32 + lpart * 16]);
      unsigned ob = (unsigned)(uintptr_t)(&lB[nb][lrow * 32 + lpart * 16]);
      async_cp16(oa, gA + k1);
      async_cp16(oa + 16, gA + k1 + 8);
      async_cp16(ob, gB + k1);
      async_cp16(ob + 16, gB + k1 + 8);
    }

    const _Float16* As = &lA[i & 1][0];
    const _Float16* Bs = &lB[i & 1][0];
    v16h bf[2];
    for (int nt = 0; nt < 2; ++nt) {
      const _Float16* p = Bs + (wn + nt * 16 + l16) * 32 + hi * 8;
      bf[nt] = ld_frag2(p, p + 16);
    }
    for (int mt = 0; mt < 4; ++mt) {
      const _Float16* p = As + (wm + mt * 16 + l16) * 32;
      v16h af = ld_frag2(p + hi * 8, p + 16 + hi * 8);
      acc[mt][0] = wmma16(af, bf[0], acc[mt][0]);
      acc[mt][1] = wmma16(af, bf[1], acc[mt][1]);
    }
  }

  // ---- LoRA epilogue: one extra WMMA (K = R = 32) per 16x16 tile ----
  v16h lbf[2];
  for (int nt = 0; nt < 2; ++nt) {
    const _Float16* p = LB + (size_t)(n0 + wn + nt * 16 + l16) * R + hi * 8;
    lbf[nt] = ld_frag2(p, p + 16);
  }
  for (int mt = 0; mt < 4; ++mt) {
    const _Float16* p = T + (size_t)(m0 + wm + mt * 16 + l16) * R;
    v16h tf = ld_frag2(p + hi * 8, p + 16 + hi * 8);
    acc[mt][0] = wmma16(tf, lbf[0], acc[mt][0]);
    acc[mt][1] = wmma16(tf, lbf[1], acc[mt][1]);
  }

  // ---- bias + (relu|residual) store ----
  for (int nt = 0; nt < 2; ++nt) {
    int col  = n0 + wn + nt * 16 + l16;
    float bv = bias[col];
    for (int mt = 0; mt < 4; ++mt) {
      for (int i = 0; i < 8; ++i) {
        int m   = m0 + wm + mt * 16 + hi * 8 + i;
        float v = acc[mt][nt][i] + bv;
        size_t idx = (size_t)m * DIM + col;
        if (do_resid) {
          H[idx] = v + H[idx];                   // pre-LN residual, fp32
        } else {
          if (do_relu) v = fmaxf(v, 0.0f);
          Out16[idx] = (_Float16)v;
        }
      }
    }
  }
}

// ---------------- fused LayerNorm: fp32 in-place + fp16 copy ----------------
__global__ __launch_bounds__(256) void ln_kernel(float* __restrict__ H,
                                                 const float* __restrict__ g,
                                                 const float* __restrict__ b,
                                                 _Float16* __restrict__ A16) {
  __shared__ float ss[256];
  __shared__ float sq[256];
  int row = blockIdx.x;
  int t   = threadIdx.x;
  float* x = H + (size_t)row * DIM;
  int c = t * 4;
  float4 v = *(const float4*)(x + c);
  ss[t] = v.x + v.y + v.z + v.w;
  sq[t] = v.x * v.x + v.y * v.y + v.z * v.z + v.w * v.w;
  __syncthreads();
  for (int off = 128; off > 0; off >>= 1) {
    if (t < off) { ss[t] += ss[t + off]; sq[t] += sq[t + off]; }
    __syncthreads();
  }
  float mu  = ss[0] * (1.0f / DIM);
  float var = sq[0] * (1.0f / DIM) - mu * mu;
  float rs  = rsqrtf(var + 1e-5f);
  float y0 = (v.x - mu) * rs * g[c + 0] + b[c + 0];
  float y1 = (v.y - mu) * rs * g[c + 1] + b[c + 1];
  float y2 = (v.z - mu) * rs * g[c + 2] + b[c + 2];
  float y3 = (v.w - mu) * rs * g[c + 3] + b[c + 3];
  float4 o = {y0, y1, y2, y3};
  *(float4*)(x + c) = o;
  v4h h = {(_Float16)y0, (_Float16)y1, (_Float16)y2, (_Float16)y3};
  *(v4h*)(A16 + (size_t)row * DIM + c) = h;
}

// ---------------------------------------------------------------------------
extern "C" void kernel_launch(void* const* d_in, const int* in_sizes, int n_in,
                              void* d_out, int out_size, void* d_ws,
                              size_t ws_size, hipStream_t stream) {
  const float* x    = (const float*)d_in[0];
  const int*   qw   = (const int*)d_in[1];
  const float* sc   = (const float*)d_in[2];
  const float* bias = (const float*)d_in[3];
  const float* la   = (const float*)d_in[4];
  const float* lb   = (const float*)d_in[5];
  const float* lng  = (const float*)d_in[6];
  const float* lnb  = (const float*)d_in[7];
  float* out = (float*)d_out;

  size_t off = 0;
  auto carve = [&](size_t bytes) {
    char* p = (char*)d_ws + off;
    off += (bytes + 255) & ~(size_t)255;
    return p;
  };
  _Float16* W16  = (_Float16*)carve((size_t)NL * DIM * DIM * 2);   // 37.75 MB
  _Float16* la16 = (_Float16*)carve((size_t)NL * R * DIM * 2);     //  1.18 MB
  _Float16* lb16 = (_Float16*)carve((size_t)NL * DIM * R * 2);     //  1.18 MB
  _Float16* t16  = (_Float16*)carve((size_t)BATCH * R * 2);        //  2.10 MB
  _Float16* actA = (_Float16*)carve((size_t)BATCH * DIM * 2);      // 67.11 MB
  _Float16* actB = (_Float16*)carve((size_t)BATCH * DIM * 2);      // 67.11 MB

  // Prologue: dequant all weights to fp16 (L2-resident thereafter), convert
  // lora mats and x, init fp32 residual buffer h = x in d_out.
  {
    int n4 = NL * DIM * DIM / 4;
    dequant_w<<<n4 / 256, 256, 0, stream>>>(qw, sc, W16, n4);
    int la4 = NL * R * DIM / 4;
    cvt_f32_f16<<<(la4 + 255) / 256, 256, 0, stream>>>(la, la16, la4);
    int lb4 = NL * DIM * R / 4;
    cvt_f32_f16<<<(lb4 + 255) / 256, 256, 0, stream>>>(lb, lb16, lb4);
    int x4 = BATCH * DIM / 4;
    cvt_f32_f16<<<x4 / 256, 256, 0, stream>>>(x, actA, x4);
    hipMemcpyAsync(out, x, (size_t)BATCH * DIM * sizeof(float),
                   hipMemcpyDeviceToDevice, stream);
  }

  dim3 ggrid(BATCH / 128, DIM / 128);
  _Float16* cur = actA;
  _Float16* oth = actB;

  for (int blk = 0; blk < 6; ++blk) {
    for (int j = 0; j < 3; ++j) {
      int li = blk * 3 + j;
      lora_t_kernel<<<BATCH / 128, 256, 0, stream>>>(
          cur, la16 + (size_t)li * R * DIM, t16);
      int resid = (j == 2) ? 1 : 0;
      int relu  = (j < 2) ? 1 : 0;
      gemm_qlora<<<ggrid, 256, 0, stream>>>(
          cur, W16 + (size_t)li * DIM * DIM, lb16 + (size_t)li * DIM * R, t16,
          bias + (size_t)li * DIM, oth, out, relu, resid);
      if (!resid) { _Float16* tmp = cur; cur = oth; oth = tmp; }
    }
    if (blk < 5) {
      ln_kernel<<<BATCH, 256, 0, stream>>>(out, lng + (size_t)blk * DIM,
                                           lnb + (size_t)blk * DIM, actA);
      cur = actA;
      oth = actB;
    }
  }
}
